// SiameseConv_52948356825901
// MI455X (gfx1250) — matrix operand order repaired and checked by reading, not compiled, not run on hardware
//
#include <hip/hip_runtime.h>

typedef __attribute__((ext_vector_type(16))) __bf16        v16bf;
typedef __attribute__((ext_vector_type(8)))  float         v8f;
typedef __attribute__((ext_vector_type(4)))  float         f32x4;
typedef __attribute__((ext_vector_type(4)))  unsigned short u16x4;
typedef __attribute__((ext_vector_type(8)))  unsigned short u16x8;
typedef __attribute__((ext_vector_type(16))) unsigned short u16x16;

#define BATCH   64
#define WD      20           // detection width/height
#define CIN     256
#define OUTC    20
#define WO      17
#define MROWS   289          // 17*17
#define MTILES  19           // ceil(289/16)
#define WAVES   8

#define SPLIT      2         // M-split: 2 workgroups per batch element
#define MTL0       10        // mt tiles handled by half 0 (half 1 gets 9)
#define DET_YROWS  13        // max detection rows staged per half (half0: 0..12, half1: 9..19)
#define DET_POS    (DET_YROWS * WD)          // 260 staged (y,x) positions
#define DET_STRIDE 264       // 256 + 8 pad halfwords -> conflict-free ds_load_b128
#define BT_STRIDE  264
#define BT_ROWS    32        // N padded 20 -> 32
#define DET_HALFS  (DET_POS * DET_STRIDE)    // 68640
#define BT_HALFS   (BT_ROWS * BT_STRIDE)     // 8448 (per buffer, double buffered)
#define SMEM_BYTES ((DET_HALFS + 2 * BT_HALFS) * 2)   // 171072 B < 320 KB WGP LDS

#define TQ      5            // template f32x4 quads per thread per chunk (5*256*4 = 5120)

__device__ __forceinline__ unsigned short f2bf(float f) {
    unsigned int u = __float_as_uint(f);
    u += 0x7FFFu + ((u >> 16) & 1u);   // round-to-nearest-even
    return (unsigned short)(u >> 16);
}

// One full K=256 pass for a single A row-tile against both N-tiles.
// No EXEC masking inside: callers branch (wave-uniformly) once per chunk.
__device__ __forceinline__ void kb_pass(const unsigned short* __restrict__ arow,
                                        const unsigned short* __restrict__ brow0,
                                        const unsigned short* __restrict__ brow1,
                                        int g, v8f& c0, v8f& c1)
{
#pragma unroll
    for (int kb = 0; kb < CIN; kb += 32) {
        // A 16x32 bf16 fragment: lane half g holds K {kb+16g .. kb+16g+15}
        u16x8 alo = *(const u16x8*)(arow + kb + 16 * g);
        u16x8 ahi = *(const u16x8*)(arow + kb + 16 * g + 8);
        u16x16 au;
#pragma unroll
        for (int e = 0; e < 8; ++e) { au[e] = alo[e]; au[e + 8] = ahi[e]; }
        // B 32x16 bf16 fragments: lane half g holds K {kb+16g .. kb+16g+15}
        u16x8 b0lo = *(const u16x8*)(brow0 + kb + 16 * g);
        u16x8 b0hi = *(const u16x8*)(brow0 + kb + 16 * g + 8);
        u16x8 b1lo = *(const u16x8*)(brow1 + kb + 16 * g);
        u16x8 b1hi = *(const u16x8*)(brow1 + kb + 16 * g + 8);
        u16x16 bu0, bu1;
#pragma unroll
        for (int e = 0; e < 8; ++e) { bu0[e] = b0lo[e]; bu0[e + 8] = b0hi[e]; }
#pragma unroll
        for (int e = 0; e < 8; ++e) { bu1[e] = b1lo[e]; bu1[e + 8] = b1hi[e]; }

        v16bf af  = __builtin_bit_cast(v16bf, au);
        v16bf bf0 = __builtin_bit_cast(v16bf, bu0);
        v16bf bf1 = __builtin_bit_cast(v16bf, bu1);
        c0 = __builtin_amdgcn_wmma_f32_16x16x32_bf16(false, af, false, bf0,
                                                     (short)0, c0, false, false);
        c1 = __builtin_amdgcn_wmma_f32_16x16x32_bf16(false, af, false, bf1,
                                                     (short)0, c1, false, false);
    }
}

__global__ __launch_bounds__(256)
void siamese_corr_wmma(const float* __restrict__ det,
                       const float* __restrict__ tmpl,
                       float* __restrict__ out)
{
    extern __shared__ unsigned short smem[];
    unsigned short* dlds = smem;               // [<=260][264] bf16 detection slice
    unsigned short* blds = smem + DET_HALFS;   // 2 x [32][264] bf16 B^T (n,k), double buffered

    const int bid  = blockIdx.x;
    const int b    = bid >> 1;
    const int half = bid & 1;
    const int tid  = threadIdx.x;
    const int lane = tid & 31;
    const int wave = tid >> 5;
    const int l15  = lane & 15;
    const int g    = lane >> 4;

    const int mt_base  = half ? MTL0 : 0;
    const int mt_count = half ? (MTILES - MTL0) : MTL0;   // 9 : 10
    const int y0       = half ? 9 : 0;
    const int yrows    = half ? 11 : 13;

    const float* tg = tmpl + (size_t)b * (16 * CIN * OUTC);

    // ---- prologue: issue chunk-0 template loads (stay in flight during det staging) ----
    f32x4 r[TQ];
#pragma unroll
    for (int q = 0; q < TQ; ++q)
        r[q] = *(const f32x4*)(tg + ((q * 256 + tid) << 2));

    // ---- stage detection slice rows [y0, y0+yrows) as bf16 (vectorized x4, coalesced) ----
    const float* dg = det + ((size_t)b * (WD * WD) + (size_t)y0 * WD) * CIN;
    const int quads = (yrows * WD * CIN) >> 2;
    for (int q = tid; q < quads; q += 256) {
        int base = q << 2;
        int pos  = base >> 8;      // /256
        int c    = base & 255;
        f32x4 v = *(const f32x4*)(dg + base);
        u16x4 h;
        h[0] = f2bf(v[0]); h[1] = f2bf(v[1]); h[2] = f2bf(v[2]); h[3] = f2bf(v[3]);
        *(u16x4*)(dlds + pos * DET_STRIDE + c) = h;
    }
    // zero N-padding rows 20..31 of both B buffers (never overwritten afterwards)
#pragma unroll
    for (int buf = 0; buf < 2; ++buf)
        for (int idx = tid; idx < (BT_ROWS - OUTC) * BT_STRIDE; idx += 256)
            blds[buf * BT_HALFS + OUTC * BT_STRIDE + idx] = 0;
    // store chunk 0 templates -> buffer 0 (transposed [o][c])
#pragma unroll
    for (int q = 0; q < TQ; ++q)
#pragma unroll
        for (int e = 0; e < 4; ++e) {
            int idx = ((q * 256 + tid) << 2) + e;
            int c = idx / OUTC;
            int o = idx - c * OUTC;
            blds[o * BT_STRIDE + c] = f2bf(r[q][e]);
        }

    // ---- per-wave tile constants: mt slots u -> mt = mt_base + wave + 8u ----
    int mtl1 = wave + WAVES;
    const bool uvalid1 = (mtl1 < mt_count);    // wave-uniform; slot 0 always valid
    int apos0, apos1;
    {
        int m = (mt_base + wave) * 16 + l15;
        if (m > MROWS - 1) m = MROWS - 1;
        int y = m / WO, x = m - y * WO;
        apos0 = ((y - y0) * WD + x) * DET_STRIDE;
    }
    {
        int mt = mt_base + (uvalid1 ? mtl1 : 0);
        int m = mt * 16 + l15;
        if (m > MROWS - 1) m = MROWS - 1;
        int y = m / WO, x = m - y * WO;
        apos1 = ((y - y0) * WD + x) * DET_STRIDE;
    }

    v8f acc00 = {}, acc01 = {}, acc10 = {}, acc11 = {};

    __syncthreads();

    for (int chunk = 0; chunk < 16; ++chunk) {
        const int cur = chunk & 1;

        // issue next chunk's template loads now; they fly during compute
        if (chunk < 15) {
            const float* tn = tg + (chunk + 1) * (CIN * OUTC);
#pragma unroll
            for (int q = 0; q < TQ; ++q)
                r[q] = *(const f32x4*)(tn + ((q * 256 + tid) << 2));
        }

        const int aoff = ((chunk >> 2) * WD + (chunk & 3)) * DET_STRIDE;
        const unsigned short* bl    = blds + cur * BT_HALFS;
        const unsigned short* brow0 = bl + l15 * BT_STRIDE;          // nt = 0
        const unsigned short* brow1 = bl + (16 + l15) * BT_STRIDE;   // nt = 1

        kb_pass(dlds + apos0 + aoff, brow0, brow1, g, acc00, acc01);
        if (uvalid1)   // one uniform branch per chunk (waves 0-1 of half0, wave 0 of half1)
            kb_pass(dlds + apos1 + aoff, brow0, brow1, g, acc10, acc11);

        // convert+store next chunk into the other buffer, then one barrier per chunk
        if (chunk < 15) {
            unsigned short* bn = blds + (1 - cur) * BT_HALFS;
#pragma unroll
            for (int q = 0; q < TQ; ++q)
#pragma unroll
                for (int e = 0; e < 4; ++e) {
                    int idx = ((q * 256 + tid) << 2) + e;
                    int c = idx / OUTC;
                    int o = idx - c * OUTC;
                    bn[o * BT_STRIDE + c] = f2bf(r[q][e]);
                }
        }
        __syncthreads();
    }

    // ---- store: VGPR rr, lane-half g -> M = 16*mt + rr + 8g, N = 16*nt + l15 ----
    float* og = out + (size_t)b * (MROWS * OUTC);
    {
        int mt = mt_base + wave;
#pragma unroll
        for (int rr = 0; rr < 8; ++rr) {
            int m = mt * 16 + rr + 8 * g;
            if (m < MROWS) {
                og[m * OUTC + l15] = acc00[rr];
                if (l15 < OUTC - 16)
                    og[m * OUTC + 16 + l15] = acc01[rr];
            }
        }
    }
    if (uvalid1) {
        int mt = mt_base + wave + WAVES;
#pragma unroll
        for (int rr = 0; rr < 8; ++rr) {
            int m = mt * 16 + rr + 8 * g;
            if (m < MROWS) {
                og[m * OUTC + l15] = acc10[rr];
                if (l15 < OUTC - 16)
                    og[m * OUTC + 16 + l15] = acc11[rr];
            }
        }
    }
}

extern "C" void kernel_launch(void* const* d_in, const int* in_sizes, int n_in,
                              void* d_out, int out_size, void* d_ws, size_t ws_size,
                              hipStream_t stream) {
    (void)in_sizes; (void)n_in; (void)out_size; (void)d_ws; (void)ws_size;
    const float* detection = (const float*)d_in[0];  // [64,20,20,256] f32
    const float* templates = (const float*)d_in[1];  // [64,4,4,5120]  f32
    float* out = (float*)d_out;                      // [64,17,17,20]  f32

    siamese_corr_wmma<<<BATCH * SPLIT, 256, SMEM_BYTES, stream>>>(detection, templates, out);
}
